// RoPEMultiHeadAttention_13915694039778
// MI455X (gfx1250) — matrix operand order repaired and checked by reading, not compiled, not run on hardware
//
#include <hip/hip_runtime.h>

typedef __bf16 bf16;
typedef __attribute__((ext_vector_type(16))) __bf16 v16bf;
typedef __attribute__((ext_vector_type(8)))  __bf16 v8bf;
typedef __attribute__((ext_vector_type(8)))  float  v8f;
typedef __attribute__((ext_vector_type(4)))  int    vi4;

constexpr int Bv = 2, Sv = 2048, Hv = 16, Dv = 64;
constexpr int Ev = Hv * Dv;       // 1024
constexpr int Mv = Bv * Sv;       // 4096

#if __has_builtin(__builtin_amdgcn_global_load_async_to_lds_b128)
#define USE_ASYNC_LDS 1
#else
#define USE_ASYNC_LDS 0
#endif

// ---------------------------------------------------------------------------
// Stage 1: casts
// ---------------------------------------------------------------------------
__global__ void cast_f32_bf16(const float* __restrict__ src, bf16* __restrict__ dst, int n) {
  int i = blockIdx.x * blockDim.x + threadIdx.x;
  if (i < n) dst[i] = (bf16)src[i];
}

// dst[n][k] = src[k][n]  (E x E), so GEMM B-fragments read contiguous K.
__global__ void cast_transpose_bf16(const float* __restrict__ src, bf16* __restrict__ dst) {
  int i = blockIdx.x * blockDim.x + threadIdx.x;   // over E*E
  const int n = i & (Ev - 1);
  const int k = i >> 10;
  dst[(size_t)n * Ev + k] = (bf16)src[(size_t)k * Ev + n];
}

// ---------------------------------------------------------------------------
// bf16 WMMA GEMM: C(MxN) = A(MxK) * B(KxN), A row-major bf16, Bt = [N][K] bf16.
// Block tile 128x64 (8 waves x 16 rows, 64 cols). B K-panel staged in LDS
// (async-to-LDS when available); each wave owns a 16x64 tile -> 4 accumulators,
// A-fragment reused 4x per WMMA group.
// ---------------------------------------------------------------------------
template <typename OutT>
__global__ void __launch_bounds__(256)
gemm_bf16_wmma(const bf16* __restrict__ A, const bf16* __restrict__ Bt,
               OutT* __restrict__ C, int M, int N, int K) {
  __shared__ bf16 bpanel[64][72];           // [nLocal][kLocal], +8 pad (16B-aligned rows)

  const int lane = threadIdx.x & 31;
  const int wid  = threadIdx.x >> 5;
  const int tilesNb = N >> 6;               // 64-wide block cols
  const int bm = blockIdx.x / tilesNb;
  const int bn = blockIdx.x % tilesNb;
  const int row0 = bm * 128 + wid * 16;
  const int col0 = bn * 64;
  const int l16 = lane & 15, hl = lane >> 4;

  v8f acc[4] = {};
  const bf16* pa_row = A + (size_t)(row0 + l16) * K;

  for (int k0 = 0; k0 < K; k0 += 64) {
    __syncthreads();                        // previous panel fully consumed
    // ---- stage Bt[col0..col0+63][k0..k0+63] into LDS (2x 16B per thread) ----
#pragma unroll
    for (int c = 0; c < 2; ++c) {
      const int chunk = threadIdx.x * 2 + c;    // 512 chunks of 8 bf16
      const int brow = chunk >> 3;              // 0..63
      const int koff = (chunk & 7) * 8;         // 0..56
      const bf16* src = Bt + (size_t)(col0 + brow) * K + k0 + koff;
#if USE_ASYNC_LDS
      __builtin_amdgcn_global_load_async_to_lds_b128(
          (vi4*)const_cast<bf16*>(src),
          (vi4*)&bpanel[brow][koff],
          0, 0);
#else
      *(v8bf*)&bpanel[brow][koff] = *(const v8bf*)src;
#endif
    }
#if USE_ASYNC_LDS
#if __has_builtin(__builtin_amdgcn_s_wait_asynccnt)
    __builtin_amdgcn_s_wait_asynccnt(0);
#else
    asm volatile("s_wait_asynccnt 0" ::: "memory");
#endif
#endif
    __syncthreads();

    // ---- compute on the panel ----
#pragma unroll
    for (int ks = 0; ks < 64; ks += 32) {
      const bf16* pa = pa_row + k0 + ks + hl * 8;
      v8bf alo = *(const v8bf*)pa;
      v8bf ahi = *(const v8bf*)(pa + 16);
      v16bf a = __builtin_shufflevector(alo, ahi, 0,1,2,3,4,5,6,7,8,9,10,11,12,13,14,15);
#pragma unroll
      for (int t2 = 0; t2 < 4; ++t2) {
        const bf16* pb = &bpanel[t2 * 16 + l16][ks + hl * 8];
        v8bf blo = *(const v8bf*)pb;
        v8bf bhi = *(const v8bf*)(pb + 16);
        v16bf b = __builtin_shufflevector(blo, bhi, 0,1,2,3,4,5,6,7,8,9,10,11,12,13,14,15);
        acc[t2] = __builtin_amdgcn_wmma_f32_16x16x32_bf16(false, a, false, b, (short)0, acc[t2], false, false);
      }
    }
  }

#pragma unroll
  for (int t2 = 0; t2 < 4; ++t2) {
    OutT* pc = C + (size_t)(row0 + hl * 8) * N + col0 + t2 * 16 + l16;
#pragma unroll
    for (int r = 0; r < 8; ++r) pc[(size_t)r * N] = (OutT)acc[t2][r];
  }
}

// ---------------------------------------------------------------------------
// Stage 3: RoPE (interleaved rotate-half variant) + repack.
//   Qo,Ko: [b][h][s][d]  (A/B fragment friendly for scores)
//   Vto:   [b][h][d][s]  (B fragment friendly for PV)
// ---------------------------------------------------------------------------
__global__ void rope_pack_kernel(const bf16* __restrict__ qraw, const bf16* __restrict__ kraw,
                                 const bf16* __restrict__ vraw, bf16* __restrict__ Qo,
                                 bf16* __restrict__ Ko, bf16* __restrict__ Vto) {
  const int idx = blockIdx.x * blockDim.x + threadIdx.x;   // over B*S*H*D = 2^22
  const int d = idx & (Dv - 1);
  const int h = (idx >> 6) & (Hv - 1);
  const int s = (idx >> 10) & (Sv - 1);
  const int b = idx >> 21;
  const size_t base = ((size_t)(b * Sv + s)) * Ev + h * Dv;

  const float xq = (float)qraw[base + d];
  const float xk = (float)kraw[base + d];
  int pd; float sign;
  if (d < 32) { pd = 2 * d + 1;     sign = -1.f; }   // rotated[j]    = -x[2j+1]
  else        { pd = 2 * (d - 32);  sign =  1.f; }   // rotated[32+j] =  x[2j]
  const float xqp = (float)qraw[base + pd];
  const float xkp = (float)kraw[base + pd];

  const int f = d & 31;
  const float invf = __powf(10000.f, -(float)f * (1.f / 32.f));
  const float ang = (float)s * invf;
  float sn, c;
  __sincosf(ang, &sn, &c);

  const size_t obase = ((size_t)(b * Hv + h) * Sv + s) * Dv + d;
  Qo[obase] = (bf16)(xq * c + sign * xqp * sn);
  Ko[obase] = (bf16)(xk * c + sign * xkp * sn);
  Vto[((size_t)(b * Hv + h) * Dv + d) * Sv + s] = vraw[base + d];
}

// ---------------------------------------------------------------------------
// Stage 4: flash attention. One wave per (b,h,16-query tile). Causal.
// ---------------------------------------------------------------------------
__global__ void __launch_bounds__(256)
attn_fa_kernel(const bf16* __restrict__ Q, const bf16* __restrict__ K,
               const bf16* __restrict__ Vt, bf16* __restrict__ O) {
  __shared__ bf16 pbuf[8][16][36];          // per-wave P staging (bf16)
  const int lane = threadIdx.x & 31;
  const int wid  = threadIdx.x >> 5;
  const int gw   = blockIdx.x * 8 + wid;
  const int qtiles = Sv / 16;               // 128
  const int bh = gw / qtiles;
  const int qt = gw % qtiles;
  const int q0 = qt * 16;
  const int b = bh / Hv, h = bh % Hv;
  const bf16* Qh = Q  + (size_t)bh * Sv * Dv;
  const bf16* Kh = K  + (size_t)bh * Sv * Dv;
  const bf16* Vh = Vt + (size_t)bh * Dv * Sv;
  const int l16 = lane & 15, hl = lane >> 4;

  // Q A-fragments (loop invariant), d-chunks [0,32) and [32,64)
  v16bf aq[2];
#pragma unroll
  for (int dc = 0; dc < 2; ++dc) {
    const bf16* pq = Qh + (size_t)(q0 + l16) * Dv + dc * 32 + hl * 16;
    v8bf lo = *(const v8bf*)pq;
    v8bf hi = *(const v8bf*)(pq + 8);
    aq[dc] = __builtin_shufflevector(lo, hi, 0,1,2,3,4,5,6,7,8,9,10,11,12,13,14,15);
  }

  v8f acc[4] = {};                          // out tiles over d = 4 x 16
  float rmax[8], rsum[8];
#pragma unroll
  for (int r = 0; r < 8; ++r) { rmax[r] = -1e30f; rsum[r] = 0.f; }

  const float kscale = 0.125f;              // 1/sqrt(64)
  const int kend = q0 + 16;                 // causal trim
  for (int kk = 0; kk < kend; kk += 32) {
    float p[2][8];
    float tmax[8];
#pragma unroll
    for (int r = 0; r < 8; ++r) tmax[r] = -1e30f;

#pragma unroll
    for (int t = 0; t < 2; ++t) {
      const int key0 = kk + t * 16;
      v8f sc = {};
#pragma unroll
      for (int dc = 0; dc < 2; ++dc) {
        const bf16* pk = Kh + (size_t)(key0 + l16) * Dv + dc * 32 + hl * 16;
        v16bf bk = *(const v16bf*)pk;
        sc = __builtin_amdgcn_wmma_f32_16x16x32_bf16(false, aq[dc], false, bk, (short)0, sc, false, false);
      }
      const int key = key0 + l16;
#pragma unroll
      for (int r = 0; r < 8; ++r) {
        const int qrow = q0 + hl * 8 + r;   // C layout: lanes0-15 rows 0-7, lanes16-31 rows 8-15
        float s = sc[r] * kscale;
        if (key > qrow) s = -1e30f;         // causal mask
        p[t][r] = s;
        tmax[r] = fmaxf(tmax[r], s);
      }
    }

    // online softmax; row state lives replicated across the 16-lane half
#pragma unroll
    for (int r = 0; r < 8; ++r) {
      float mx = tmax[r];
      mx = fmaxf(mx, __shfl_xor(mx, 1, 32));
      mx = fmaxf(mx, __shfl_xor(mx, 2, 32));
      mx = fmaxf(mx, __shfl_xor(mx, 4, 32));
      mx = fmaxf(mx, __shfl_xor(mx, 8, 32));
      const float nm = fmaxf(rmax[r], mx);
      const float sf = __expf(rmax[r] - nm);
      rmax[r] = nm;
      const float p0 = __expf(p[0][r] - nm);
      const float p1 = __expf(p[1][r] - nm);
      float rs = p0 + p1;
      rs += __shfl_xor(rs, 1, 32);
      rs += __shfl_xor(rs, 2, 32);
      rs += __shfl_xor(rs, 4, 32);
      rs += __shfl_xor(rs, 8, 32);
      rsum[r] = rsum[r] * sf + rs;
#pragma unroll
      for (int t2 = 0; t2 < 4; ++t2) acc[t2][r] *= sf;
      const int row = hl * 8 + r;
      pbuf[wid][row][l16]      = (bf16)p0;
      pbuf[wid][row][16 + l16] = (bf16)p1;
    }
    __builtin_amdgcn_wave_barrier();
    asm volatile("s_wait_dscnt 0" ::: "memory");   // C-layout -> A-layout via LDS, same wave

    v16bf ap;
#pragma unroll
    for (int i = 0; i < 8; ++i) {
      ap[i]     = pbuf[wid][l16][hl * 16 + i];
      ap[8 + i] = pbuf[wid][l16][hl * 16 + 8 + i];
    }

#pragma unroll
    for (int t2 = 0; t2 < 4; ++t2) {
      const bf16* pv = Vh + (size_t)(t2 * 16 + l16) * Sv + kk + hl * 16;
      v16bf bv = *(const v16bf*)pv;
      acc[t2] = __builtin_amdgcn_wmma_f32_16x16x32_bf16(false, ap, false, bv, (short)0, acc[t2], false, false);
    }
    __builtin_amdgcn_wave_barrier();
  }

  // normalize + store bf16 [b][s][e] for the final projection GEMM
#pragma unroll
  for (int t2 = 0; t2 < 4; ++t2) {
#pragma unroll
    for (int r = 0; r < 8; ++r) {
      const int row = hl * 8 + r;
      const int e = h * Dv + t2 * 16 + l16;
      O[(size_t)(b * Sv + q0 + row) * Ev + e] = (bf16)(acc[t2][r] / rsum[r]);
    }
  }
}

// ---------------------------------------------------------------------------
extern "C" void kernel_launch(void* const* d_in, const int* in_sizes, int n_in,
                              void* d_out, int out_size, void* d_ws, size_t ws_size,
                              hipStream_t stream) {
  (void)in_sizes; (void)n_in; (void)out_size; (void)ws_size;
  const float* x  = (const float*)d_in[0];
  // d_in[1] = causal mask (bool) — structure known, unused
  const float* Wq = (const float*)d_in[2];
  const float* Wk = (const float*)d_in[3];
  const float* Wv = (const float*)d_in[4];
  const float* Wo = (const float*)d_in[5];
  float* out = (float*)d_out;

  char* ws = (char*)d_ws;
  size_t off = 0;
  auto alloc = [&](size_t elems) { bf16* p = (bf16*)(ws + off); off += elems * sizeof(bf16); return p; };
  bf16* xbf  = alloc((size_t)Mv * Ev);
  bf16* wqT  = alloc((size_t)Ev * Ev);
  bf16* wkT  = alloc((size_t)Ev * Ev);
  bf16* wvT  = alloc((size_t)Ev * Ev);
  bf16* woT  = alloc((size_t)Ev * Ev);
  bf16* qraw = alloc((size_t)Mv * Ev);
  bf16* kraw = alloc((size_t)Mv * Ev);
  bf16* vraw = alloc((size_t)Mv * Ev);
  bf16* Qbf  = alloc((size_t)Mv * Ev);   // [b][h][s][d]
  bf16* Kbf  = alloc((size_t)Mv * Ev);   // [b][h][s][d]
  bf16* Vtb  = alloc((size_t)Mv * Ev);   // [b][h][d][s]
  bf16* obf  = alloc((size_t)Mv * Ev);   // [b][s][e]

  const int tpb = 256;
  cast_f32_bf16<<<(Mv * Ev) / tpb, tpb, 0, stream>>>(x, xbf, Mv * Ev);
  cast_transpose_bf16<<<(Ev * Ev) / tpb, tpb, 0, stream>>>(Wq, wqT);
  cast_transpose_bf16<<<(Ev * Ev) / tpb, tpb, 0, stream>>>(Wk, wkT);
  cast_transpose_bf16<<<(Ev * Ev) / tpb, tpb, 0, stream>>>(Wv, wvT);
  cast_transpose_bf16<<<(Ev * Ev) / tpb, tpb, 0, stream>>>(Wo, woT);

  const int gemmBlocks = (Mv / 128) * (Ev / 64);   // 128x64 block tiles, 8 waves
  gemm_bf16_wmma<bf16><<<gemmBlocks, tpb, 0, stream>>>(xbf, wqT, qraw, Mv, Ev, Ev);
  gemm_bf16_wmma<bf16><<<gemmBlocks, tpb, 0, stream>>>(xbf, wkT, kraw, Mv, Ev, Ev);
  gemm_bf16_wmma<bf16><<<gemmBlocks, tpb, 0, stream>>>(xbf, wvT, vraw, Mv, Ev, Ev);

  rope_pack_kernel<<<(Bv * Sv * Hv * Dv) / tpb, tpb, 0, stream>>>(qraw, kraw, vraw, Qbf, Kbf, Vtb);

  attn_fa_kernel<<<(Bv * Hv * (Sv / 16)) / 8, tpb, 0, stream>>>(Qbf, Kbf, Vtb, obf);

  gemm_bf16_wmma<float><<<gemmBlocks, tpb, 0, stream>>>(obf, woT, out, Mv, Ev, Ev);
}
